// Transducer_26577257628143
// MI455X (gfx1250) — compile-verified, matrix operand order checked by reading, not run on hardware
//
#include <hip/hip_runtime.h>
#include <math.h>

// ---------------- types ----------------
typedef __bf16 bf16_t;
typedef __attribute__((ext_vector_type(16))) __bf16 v16bf;
typedef __attribute__((ext_vector_type(8)))  float  v8f;
typedef __attribute__((ext_vector_type(4)))  float  v4f;

union ABu { v16bf v; v4f f[2]; };

// problem constants
#define BB   8
#define TT   150
#define U1   81
#define HH   512
#define VV   1000
#define EE   999
#define BT   (BB*TT)      // 1200
#define BU   (BB*U1)      // 648
#define ROWS (BT*U1)      // 97200

__device__ inline float sigf(float x) { return 1.0f / (1.0f + expf(-x)); }

// Load a 16-element fragment: two contiguous 8-halfword chunks.
__device__ inline v16bf load_frag(const bf16_t* p0, const bf16_t* p1) {
    ABu u;
    u.f[0] = *(const v4f*)p0;
    u.f[1] = *(const v4f*)p1;
    return u.v;
}
__device__ inline v8f wmma_bf16(v16bf a, v16bf b, v8f c) {
    return __builtin_amdgcn_wmma_f32_16x16x32_bf16(false, a, false, b, (short)0, c, false, false);
}

// ---------------- pack weights into WMMA B-fragment layout ----------------
// per (kt,nt) tile: 32 lanes x 16 contiguous bf16.
// k mapping: lane<16 -> K {0..7,16..23}; lane>=16 -> K {8..15,24..31}; n = nt*16+(lane&15)
// transB=1: B[k][n] = W[n*ldw + k]   (PyTorch (N,K) weight)
// transB=0: B[k][n] = W[k*ldw + n]
__global__ void pack_b_kernel(const float* __restrict__ W, bf16_t* __restrict__ P,
                              int K, int N, int ldw, int transB,
                              int Ktiles, int Ntiles) {
    int idx = blockIdx.x * blockDim.x + threadIdx.x;
    int total = Ktiles * Ntiles * 512;
    if (idx >= total) return;
    int e    = idx & 15;
    int lane = (idx >> 4) & 31;
    int tile = idx >> 9;
    int nt = tile % Ntiles;
    int kt = tile / Ntiles;
    int half = lane >> 4;
    int kk = (e & 7) + ((e >> 3) << 4) + (half << 3);
    int k = kt * 32 + kk;
    int n = nt * 16 + (lane & 15);
    float v = 0.0f;
    if (k < K && n < N) v = transB ? W[(size_t)n * ldw + k] : W[(size_t)k * ldw + n];
    P[idx] = (bf16_t)v;
}

// ---------------- generic WMMA GEMM: C = A(f32,MxK) * Bpacked + bias ----------------
// 64 rows x 64 cols per block; 4 waves, each owns one 16-col n-tile and
// shares its B fragment across the 4 row-tiles (4x L2 B-traffic reduction).
// grid = (ceil(M/64), ceil(N/64)); block = 128; dyn LDS = 64*Kpad*2
__global__ __launch_bounds__(128)
void wmma_gemm_kernel(const float* __restrict__ A, int lda,
                      const bf16_t* __restrict__ Bp,
                      const float* __restrict__ bias,
                      float* __restrict__ C, int ldc,
                      int M, int N, int K, int Kpad, int Ntiles) {
    extern __shared__ char smemG[];
    bf16_t* sA = (bf16_t*)smemG;            // [64][Kpad]
    int m0 = blockIdx.x * 64;
    int tid = threadIdx.x;
    for (int r = 0; r < 64; ++r) {
        int row = m0 + r;
        for (int k = tid; k < Kpad; k += 128) {
            float v = (row < M && k < K) ? A[(size_t)row * lda + k] : 0.0f;
            sA[r * Kpad + k] = (bf16_t)v;
        }
    }
    __syncthreads();
    int wave = tid >> 5, lane = tid & 31;
    int nt = blockIdx.y * 4 + wave;
    int n0 = nt * 16;
    if (n0 < N) {
        const bf16_t* abase = sA + (lane & 15) * Kpad + ((lane >> 4) << 3);
        int rstride = 16 * Kpad;
        v8f acc0 = {}, acc1 = {}, acc2 = {}, acc3 = {};
        int Ktiles = Kpad >> 5;
        for (int kt = 0; kt < Ktiles; ++kt) {
            int k0 = kt << 5;
            const bf16_t* bptr = Bp + ((((size_t)kt * Ntiles + nt) * 32 + lane) << 4);
            v16bf bf = load_frag(bptr, bptr + 8);
            v16bf a0 = load_frag(abase + k0, abase + k0 + 16);
            acc0 = wmma_bf16(a0, bf, acc0);
            v16bf a1 = load_frag(abase + rstride + k0, abase + rstride + k0 + 16);
            acc1 = wmma_bf16(a1, bf, acc1);
            v16bf a2 = load_frag(abase + 2 * rstride + k0, abase + 2 * rstride + k0 + 16);
            acc2 = wmma_bf16(a2, bf, acc2);
            v16bf a3 = load_frag(abase + 3 * rstride + k0, abase + 3 * rstride + k0 + 16);
            acc3 = wmma_bf16(a3, bf, acc3);
        }
        int col = n0 + (lane & 15);
        int rowBase = (lane >> 4) << 3;
        if (col < N) {
            float bv = bias ? bias[col] : 0.0f;
            v8f accs[4] = {acc0, acc1, acc2, acc3};
            for (int rt = 0; rt < 4; ++rt) {
                for (int i = 0; i < 8; ++i) {
                    int row = m0 + rt * 16 + rowBase + i;
                    if (row < M) C[(size_t)row * ldc + col] = accs[rt][i] + bv;
                }
            }
        }
    }
}

// ---------------- bias sum ----------------
__global__ void bias_sum_kernel(const float* __restrict__ a, const float* __restrict__ b,
                                float* __restrict__ o, int n) {
    int i = blockIdx.x * blockDim.x + threadIdx.x;
    if (i < n) o[i] = a[i] + b[i];
}

// ---------------- embedding gather (with blank pad) ----------------
__global__ void embed_kernel(const int* __restrict__ y, const float* __restrict__ emb,
                             float* __restrict__ e) {
    int idx = blockIdx.x * blockDim.x + threadIdx.x;
    if (idx >= BU * EE) return;
    int row = idx / EE, k = idx % EE;
    int b = row / U1, u = row % U1;
    int tok = (u == 0) ? 0 : y[b * (U1 - 1) + (u - 1)];
    e[idx] = emb[(size_t)tok * EE + k];
}

// ---------------- sequential LSTM scan ----------------
// grid.x selects one of two independent jobs (fwd / bwd direction) so both
// run concurrently on separate WGPs. block = 1024 (32 waves).
// preact: (B*T, 2048) f32, gate order i,f,g,o (includes x@Wih.T + biases)
// WhhP: packed bf16, K=512 (16 ktiles), N=2048 (128 ntiles)
__global__ __launch_bounds__(1024, 1)
void lstm_scan_kernel(const float* __restrict__ preact0, const bf16_t* __restrict__ whh0,
                      float* __restrict__ hout0, int rev0,
                      const float* __restrict__ preact1, const bf16_t* __restrict__ whh1,
                      float* __restrict__ hout1, int rev1,
                      int ldh, int T) {
    const float* preact; const bf16_t* WhhP; float* hout; int reverse;
    if (blockIdx.x == 0) { preact = preact0; WhhP = whh0; hout = hout0; reverse = rev0; }
    else                 { preact = preact1; WhhP = whh1; hout = hout1; reverse = rev1; }

    extern __shared__ char smem[];
    bf16_t* hA  = (bf16_t*)smem;                                // [16][512] (rows 8..15 = 0)
    float*  cst = (float*)(smem + 16 * 512 * 2);                // [8][512]
    float*  gbuf = (float*)(smem + 16 * 512 * 2 + 8 * 512 * 4); // [8][2048]
    int tid = threadIdx.x;
    for (int i = tid; i < 16 * 512; i += 1024) hA[i] = (bf16_t)0.0f;
    for (int i = tid; i < 8 * 512;  i += 1024) cst[i] = 0.0f;
    __syncthreads();
    int wave = tid >> 5, lane = tid & 31;
    const bf16_t* arow = hA + (lane & 15) * 512 + ((lane >> 4) << 3);
    for (int step = 0; step < T; ++step) {
        int t = reverse ? (T - 1 - step) : step;
        // gates += h @ Whh.T : each wave computes 4 tiles of 16 columns
        for (int j = 0; j < 4; ++j) {
            int nt = wave * 4 + j;
            int n0 = nt * 16;
            v8f acc = {};
            for (int kt = 0; kt < 16; ++kt) {
                int k0 = kt << 5;
                v16bf af = load_frag(arow + k0, arow + k0 + 16);
                const bf16_t* bptr = WhhP + ((((size_t)kt * 128 + nt) * 32 + lane) << 4);
                v16bf bf = load_frag(bptr, bptr + 8);
                acc = wmma_bf16(af, bf, acc);
            }
            if (lane < 16) {            // rows 0..7 = batch; rows 8..15 are zero-pad
                int col = n0 + lane;
                for (int i = 0; i < 8; ++i) gbuf[i * 2048 + col] = acc[i];
            }
        }
        __syncthreads();
        // pointwise LSTM cell: 8*512 (b,h) pairs over 1024 threads
        for (int j = 0; j < 4; ++j) {
            int idx = tid + j * 1024;
            int b = idx >> 9, hh = idx & 511;
            size_t pa = ((size_t)(b * T + t)) * 2048;
            float gi = gbuf[b * 2048 + hh]        + preact[pa + hh];
            float gf = gbuf[b * 2048 + 512 + hh]  + preact[pa + 512 + hh];
            float gg = gbuf[b * 2048 + 1024 + hh] + preact[pa + 1024 + hh];
            float go = gbuf[b * 2048 + 1536 + hh] + preact[pa + 1536 + hh];
            float c = sigf(gf) * cst[b * 512 + hh] + sigf(gi) * tanhf(gg);
            float h = sigf(go) * tanhf(c);
            cst[b * 512 + hh] = c;
            hA[b * 512 + hh] = (bf16_t)h;
            hout[((size_t)(b * T + t)) * ldh + hh] = h;
        }
        __syncthreads();
    }
}

// ---------------- fused joint: tanh(je+jd+jb) @ out_W + out_b -> logits ----------------
// 64 rows x 64 cols per block; each wave owns one 16-col n-tile and shares its
// B fragment across 4 row-tiles. grid = (ceil(97200/64)=1519, 16); block = 128.
// dyn LDS = 64*512*2 = 64KB
__global__ __launch_bounds__(128)
void joint_out_kernel(const float* __restrict__ je,   // (B*T, 512)
                      const float* __restrict__ jd,   // (B*U1, 512)
                      const float* __restrict__ jb,   // 512
                      const bf16_t* __restrict__ WoP, // packed K=512, Ntiles=63
                      const float* __restrict__ ob,   // 1000
                      float* __restrict__ out) {      // (97200, 1000)
    extern __shared__ char smemJ[];
    bf16_t* sJ = (bf16_t*)smemJ;            // [64][512]
    int r0 = blockIdx.x * 64;
    int tid = threadIdx.x;
    for (int i = tid; i < 64 * 512; i += 128) {
        int r = i >> 9, k = i & 511;
        int row = r0 + r;
        float v = 0.0f;
        if (row < ROWS) {
            int b = row / (TT * U1);
            int rem = row % (TT * U1);
            int t = rem / U1, u = rem % U1;
            v = tanhf(je[((size_t)(b * TT + t)) * 512 + k] +
                      jd[((size_t)(b * U1 + u)) * 512 + k] + jb[k]);
        }
        sJ[i] = (bf16_t)v;
    }
    __syncthreads();
    int wave = tid >> 5, lane = tid & 31;
    int nt = blockIdx.y * 4 + wave;
    int n0 = nt * 16;
    if (n0 < VV) {
        const bf16_t* abase = sJ + (lane & 15) * 512 + ((lane >> 4) << 3);
        const int rstride = 16 * 512;
        v8f acc0 = {}, acc1 = {}, acc2 = {}, acc3 = {};
        for (int kt = 0; kt < 16; ++kt) {
            int k0 = kt << 5;
            const bf16_t* bptr = WoP + ((((size_t)kt * 63 + nt) * 32 + lane) << 4);
            v16bf bf = load_frag(bptr, bptr + 8);
            v16bf a0 = load_frag(abase + k0, abase + k0 + 16);
            acc0 = wmma_bf16(a0, bf, acc0);
            v16bf a1 = load_frag(abase + rstride + k0, abase + rstride + k0 + 16);
            acc1 = wmma_bf16(a1, bf, acc1);
            v16bf a2 = load_frag(abase + 2 * rstride + k0, abase + 2 * rstride + k0 + 16);
            acc2 = wmma_bf16(a2, bf, acc2);
            v16bf a3 = load_frag(abase + 3 * rstride + k0, abase + 3 * rstride + k0 + 16);
            acc3 = wmma_bf16(a3, bf, acc3);
        }
        int col = n0 + (lane & 15);
        int rowBase = (lane >> 4) << 3;
        if (col < VV) {
            float bv = ob[col];
            v8f accs[4] = {acc0, acc1, acc2, acc3};
            for (int rt = 0; rt < 4; ++rt) {
                for (int i = 0; i < 8; ++i) {
                    int row = r0 + rt * 16 + rowBase + i;
                    if (row < ROWS) out[(size_t)row * VV + col] = accs[rt][i] + bv;
                }
            }
        }
    }
}

// ---------------- log_softmax over last dim (in place) ----------------
__global__ __launch_bounds__(256)
void logsoftmax_kernel(float* __restrict__ out) {
    __shared__ float red[256];
    size_t row = blockIdx.x;
    float* p = out + row * VV;
    int tid = threadIdx.x;
    float vals[4];
    float m = -INFINITY;
    for (int j = 0; j < 4; ++j) {
        int c = tid + j * 256;
        vals[j] = (c < VV) ? p[c] : -INFINITY;
        m = fmaxf(m, vals[j]);
    }
    red[tid] = m; __syncthreads();
    for (int s = 128; s > 0; s >>= 1) { if (tid < s) red[tid] = fmaxf(red[tid], red[tid + s]); __syncthreads(); }
    m = red[0]; __syncthreads();
    float s = 0.0f;
    for (int j = 0; j < 4; ++j) { if (tid + j * 256 < VV) s += expf(vals[j] - m); }
    red[tid] = s; __syncthreads();
    for (int st = 128; st > 0; st >>= 1) { if (tid < st) red[tid] += red[tid + st]; __syncthreads(); }
    float lse = logf(red[0]) + m;
    for (int j = 0; j < 4; ++j) { int c = tid + j * 256; if (c < VV) p[c] = vals[j] - lse; }
}

// ---------------- host orchestration ----------------
extern "C" void kernel_launch(void* const* d_in, const int* in_sizes, int n_in,
                              void* d_out, int out_size, void* d_ws, size_t ws_size,
                              hipStream_t stream) {
    (void)in_sizes; (void)n_in; (void)out_size; (void)ws_size;
    const float* x   = (const float*)d_in[0];
    const int*   y   = (const int*)d_in[1];
    // enc_params: idx 2 + l*8 + dir*4 + {Wih,Whh,bih,bhh}
    const float* enc_out_W = (const float*)d_in[34];
    const float* enc_out_b = (const float*)d_in[35];
    const float* emb       = (const float*)d_in[36];
    const float* dWih      = (const float*)d_in[37];
    const float* dWhh      = (const float*)d_in[38];
    const float* dbih      = (const float*)d_in[39];
    const float* dbhh      = (const float*)d_in[40];
    const float* dec_out_W = (const float*)d_in[41];
    const float* dec_out_b = (const float*)d_in[42];
    const float* joint_W   = (const float*)d_in[43];
    const float* joint_b   = (const float*)d_in[44];
    const float* out_W     = (const float*)d_in[45];
    const float* out_b     = (const float*)d_in[46];

    char* ws = (char*)d_ws;
    size_t off = 0;
    auto take = [&](size_t bytes) -> char* {
        char* p = ws + off;
        off += (bytes + 255) & ~(size_t)255;
        return p;
    };
    bf16_t* packM  = (bf16_t*)take((size_t)2097152 * 2);  // Wih / general B packs
    bf16_t* packF  = (bf16_t*)take((size_t)1048576 * 2);  // Whh fwd (16*128*512)
    bf16_t* packR  = (bf16_t*)take((size_t)1048576 * 2);  // Whh bwd
    float* preactF = (float*)take((size_t)BT * 2048 * 4);
    float* preactR = (float*)take((size_t)BT * 2048 * 4);
    float* C0      = (float*)take((size_t)BT * 1024 * 4);
    float* C1      = (float*)take((size_t)BT * 1024 * 4);
    float* encB    = (float*)take((size_t)BT * HH * 4);
    float* ebuf    = (float*)take((size_t)BU * EE * 4);
    float* dbuf    = (float*)take((size_t)BU * HH * 4);
    float* decP    = (float*)take((size_t)BU * HH * 4);
    float* je      = (float*)take((size_t)BT * HH * 4);
    float* jd      = (float*)take((size_t)BU * HH * 4);
    float* bsum    = (float*)take((size_t)2048 * 4);

    auto packB = [&](bf16_t* dst, const float* W, int K, int N, int ldw, int transB) {
        int Kt = (K + 31) / 32, Nt = (N + 15) / 16;
        int total = Kt * Nt * 512;
        pack_b_kernel<<<(total + 255) / 256, 256, 0, stream>>>(W, dst, K, N, ldw, transB, Kt, Nt);
    };
    auto gemm = [&](const float* A, int lda, const bf16_t* Bp, const float* bias,
                    float* C, int ldc, int M, int N, int K) {
        int Kpad = (K + 31) / 32 * 32;
        int Nt = (N + 15) / 16;
        dim3 grid((M + 63) / 64, (N + 63) / 64);
        size_t lds = (size_t)64 * Kpad * 2;
        wmma_gemm_kernel<<<grid, 128, lds, stream>>>(A, lda, Bp, bias, C, ldc, M, N, K, Kpad, Nt);
    };

    // ---- encoder: 4 bidirectional LSTM layers, fwd+bwd scans run concurrently ----
    const float* Ain = x;
    int lda = 80, K = 80;
    float* Cbufs[2] = {C0, C1};
    int cur = 0;
    for (int l = 0; l < 4; ++l) {
        float* Cout = Cbufs[cur];
        int bF = 2 + l * 8;          // fwd params
        int bR = 2 + l * 8 + 4;      // bwd params
        // fwd input projection
        bias_sum_kernel<<<8, 256, 0, stream>>>((const float*)d_in[bF + 2], (const float*)d_in[bF + 3], bsum, 2048);
        packB(packM, (const float*)d_in[bF + 0], K, 2048, K, 1);
        gemm(Ain, lda, packM, bsum, preactF, 2048, BT, 2048, K);
        // bwd input projection
        bias_sum_kernel<<<8, 256, 0, stream>>>((const float*)d_in[bR + 2], (const float*)d_in[bR + 3], bsum, 2048);
        packB(packM, (const float*)d_in[bR + 0], K, 2048, K, 1);
        gemm(Ain, lda, packM, bsum, preactR, 2048, BT, 2048, K);
        // recurrent weights
        packB(packF, (const float*)d_in[bF + 1], 512, 2048, 512, 1);
        packB(packR, (const float*)d_in[bR + 1], 512, 2048, 512, 1);
        // both directions concurrently (2 blocks -> 2 WGPs)
        lstm_scan_kernel<<<2, 1024, 98304, stream>>>(preactF, packF, Cout,       0,
                                                     preactR, packR, Cout + 512, 1,
                                                     1024, TT);
        Ain = Cout; lda = 1024; K = 1024;
        cur ^= 1;
    }
    // enc = concat @ enc_out_W + enc_out_b
    packB(packM, enc_out_W, 1024, 512, 512, 0);
    gemm(Ain, 1024, packM, enc_out_b, encB, 512, BT, 512, 1024);

    // ---- decoder ----
    embed_kernel<<<(BU * EE + 255) / 256, 256, 0, stream>>>(y, emb, ebuf);
    bias_sum_kernel<<<8, 256, 0, stream>>>(dbih, dbhh, bsum, 2048);
    packB(packM, dWih, EE, 2048, EE, 1);
    gemm(ebuf, EE, packM, bsum, preactF, 2048, BU, 2048, EE);
    packB(packF, dWhh, 512, 2048, 512, 1);
    lstm_scan_kernel<<<1, 1024, 98304, stream>>>(preactF, packF, dbuf, 0,
                                                 preactF, packF, dbuf, 0,
                                                 512, U1);
    packB(packM, dec_out_W, 512, 512, 512, 0);
    gemm(dbuf, 512, packM, dec_out_b, decP, 512, BU, 512, 512);

    // ---- joint projections ----
    packB(packM, joint_W, 512, 512, 512, 0);                   // rows [0:512)
    gemm(encB, 512, packM, nullptr, je, 512, BT, 512, 512);
    packB(packM, joint_W + (size_t)512 * 512, 512, 512, 512, 0); // rows [512:1024)
    gemm(decP, 512, packM, nullptr, jd, 512, BU, 512, 512);

    // ---- fused tanh-joint + output GEMM -> logits ----
    packB(packM, out_W, 512, 1000, 1000, 0);                   // Ntiles = 63
    dim3 jg((ROWS + 63) / 64, 16);
    joint_out_kernel<<<jg, 128, 65536, stream>>>(je, jd, joint_b, packM, out_b, (float*)d_out);

    // ---- log_softmax in place ----
    logsoftmax_kernel<<<ROWS, 256, 0, stream>>>((float*)d_out);
}
